// LinAttention_89739046683495
// MI455X (gfx1250) — matrix operand order repaired and checked by reading, not compiled, not run on hardware
//
#include <hip/hip_runtime.h>

// ---------------------------------------------------------------------------
// Linformer attention for MI455X (gfx1250): all contractions via
// v_wmma_f32_16x16x32_bf16 (wave32).  Software-pipelined GEMMs (ping-pong LDS,
// 1 barrier / K-step), fused scores/softmax/AV attention kernel with
// global_load_async_to_lds_b128 staging for the V operand.
// ---------------------------------------------------------------------------

#define BB 4
#define DD 1024
#define NN 4096
#define HH 16
#define KK 256
#define HD 64

typedef __attribute__((ext_vector_type(16))) __bf16 bf16x16;
typedef __attribute__((ext_vector_type(8)))  __bf16 bf16x8;
typedef __attribute__((ext_vector_type(4)))  __bf16 bf16x4;
typedef __attribute__((ext_vector_type(2)))  __bf16 bf16x2;
typedef __attribute__((ext_vector_type(8)))  float  f32x8;

static __device__ __forceinline__ __bf16 f2bf(float f) {
  // round-to-nearest-even fp32 -> bf16
  unsigned u = __builtin_bit_cast(unsigned, f);
  unsigned r = u + 0x7fffu + ((u >> 16) & 1u);
  unsigned short h = (unsigned short)(r >> 16);
  return __builtin_bit_cast(__bf16, h);
}

static __device__ __forceinline__ void cvt2(float x, float y, __bf16* o) {
#if __has_builtin(__builtin_amdgcn_cvt_pk_bf16_f32)
  bf16x2 p = __builtin_amdgcn_cvt_pk_bf16_f32(x, y);
  o[0] = p.x;
  o[1] = p.y;
#else
  o[0] = f2bf(x);
  o[1] = f2bf(y);
#endif
}

static __device__ __forceinline__ f32x8 wmma_bf16(bf16x16 a, bf16x16 b, f32x8 c) {
  // D = A(16x32) * B(32x16) + C, fp32 accumulate
  return __builtin_amdgcn_wmma_f32_16x16x32_bf16(false, a, false, b, (short)0, c,
                                                 false, false);
}

// A-fragment (16x32 bf16): lane m = l%16 holds row m; elements 0-15 = K in
// [khalf*16, khalf*16+16), khalf=l/16.
static __device__ __forceinline__ bf16x16 frag_a(const __bf16* row, int khalf) {
  return *(const bf16x16*)(row + khalf * 16);
}

// ---- staged-load helpers (convert fp32 -> bf16 on the way to LDS) ---------
static __device__ __forceinline__ void ld4(const float* p, __bf16* o) {
  const float4 v = *(const float4*)p;
  cvt2(v.x, v.y, o);
  cvt2(v.z, v.w, o + 2);
}
static __device__ __forceinline__ void ld4(const __bf16* p, __bf16* o) {
  const bf16x4 v = *(const bf16x4*)p;
  o[0] = v.x; o[1] = v.y; o[2] = v.z; o[3] = v.w;
}
static __device__ __forceinline__ void stc(float* p, float v)  { *p = v; }
static __device__ __forceinline__ void stc(__bf16* p, float v) { *p = f2bf(v); }

// ---------------------------------------------------------------------------
// Tiled WMMA GEMM: C[b] = A[b] (MxKd, row-major) * B[b] (KdxNcol, row-major)
// [+ bias[m]].  Block tile 128x128, K-step 32, 8 waves of 32x64.
// Software pipeline: next tile prefetched into registers one iteration ahead,
// ping-pong LDS buffers, single __syncthreads per K-step.
// B is staged TRANSPOSED in LDS so B-fragments are contiguous ds_load_b128s.
// ---------------------------------------------------------------------------
template <typename TA, typename TB, typename TO, bool BIAS>
__global__ __launch_bounds__(256) void gemm_wmma_kernel(
    const TA* __restrict__ A, long long A_bs,
    const TB* __restrict__ Bm, long long B_bs,
    const float* __restrict__ bias,
    TO* __restrict__ C, long long C_bs,
    int M, int Kd, int Ncol) {
  __shared__ __bf16 As[2][128 * 32];   // [m][k]
  __shared__ __bf16 Bst[2][128 * 32];  // [n][k]  (transposed stage)

  const int bz = blockIdx.z;
  A  += (size_t)A_bs * bz;
  Bm += (size_t)B_bs * bz;
  C  += (size_t)C_bs * bz;

  const int mb = blockIdx.x * 128;
  const int nb = blockIdx.y * 128;
  const int tid   = threadIdx.x;
  const int lane  = tid & 31;
  const int wave  = tid >> 5;
  const int wm    = (wave >> 1) * 32;  // 4 waves along M
  const int wn    = (wave & 1) * 64;   // 2 waves along N
  const int lm    = lane & 15;
  const int khalf = lane >> 4;

  // per-thread stage coordinates (4 quads for A, 4 quads for B)
  int arow[4], ac4[4], bkr[4], bc4[4];
#pragma unroll
  for (int i = 0; i < 4; ++i) {
    const int idx = tid + i * 256;   // 1024 quads per tile
    arow[i] = idx >> 3;
    ac4[i]  = (idx & 7) << 2;
    bkr[i]  = idx >> 5;
    bc4[i]  = (idx & 31) << 2;
  }

  __bf16 ra[16], rb[16];  // register-staged next tile (already bf16)

  auto load_tile = [&](int kb) {
#pragma unroll
    for (int i = 0; i < 4; ++i)
      ld4(A + (size_t)(mb + arow[i]) * Kd + kb + ac4[i], ra + i * 4);
#pragma unroll
    for (int i = 0; i < 4; ++i)
      ld4(Bm + (size_t)(kb + bkr[i]) * Ncol + nb + bc4[i], rb + i * 4);
  };
  auto store_tile = [&](int buf) {
#pragma unroll
    for (int i = 0; i < 4; ++i)
#pragma unroll
      for (int j = 0; j < 4; ++j) As[buf][arow[i] * 32 + ac4[i] + j] = ra[i * 4 + j];
#pragma unroll
    for (int i = 0; i < 4; ++i)
#pragma unroll
      for (int j = 0; j < 4; ++j) Bst[buf][(bc4[i] + j) * 32 + bkr[i]] = rb[i * 4 + j];
  };

  f32x8 acc[2][4] = {};

  const int nsteps = Kd >> 5;
  load_tile(0);
  store_tile(0);
  if (nsteps > 1) load_tile(32);
  __syncthreads();

  for (int s = 0; s < nsteps; ++s) {
    const int cur = s & 1;
    if (s + 1 < nsteps) store_tile(1 - cur);         // drain tile s+1
    if (s + 2 < nsteps) load_tile((s + 2) << 5);     // prefetch tile s+2

    bf16x16 af[2];
#pragma unroll
    for (int mt = 0; mt < 2; ++mt)
      af[mt] = frag_a(&As[cur][(wm + mt * 16 + lm) * 32], khalf);
#pragma unroll
    for (int nt = 0; nt < 4; ++nt) {
      bf16x16 bfr =
          *(const bf16x16*)&Bst[cur][(wn + nt * 16 + lm) * 32 + khalf * 16];
#pragma unroll
      for (int mt = 0; mt < 2; ++mt)
        acc[mt][nt] = wmma_bf16(af[mt], bfr, acc[mt][nt]);
    }
    __syncthreads();
  }

  // epilogue: C layout — lane col = l%16, VGPR r holds row r + 8*(l/16)
#pragma unroll
  for (int mt = 0; mt < 2; ++mt)
#pragma unroll
    for (int nt = 0; nt < 4; ++nt) {
      const int n = nb + wn + nt * 16 + lm;
#pragma unroll
      for (int r = 0; r < 8; ++r) {
        const int m = mb + wm + mt * 16 + khalf * 8 + r;
        float v = acc[mt][nt][r];
        if constexpr (BIAS) v += bias[m];
        stc(&C[(size_t)m * Ncol + n], v);
      }
    }
}

// ---------------------------------------------------------------------------
// Fused attention: one WG (4 waves) per (b, h, 64-row n-block).
//   scores[16x256] per wave via WMMA (contraction hd=64, 2 K-steps),
//   softmax across lanes (wave32 half-wave shfl_xor width 16),
//   prob -> LDS bf16 -> A-fragments for AV WMMA (contraction katt=256).
// V operand staged with GLOBAL_LOAD_ASYNC_TO_LDS_B128 (ASYNCcnt path);
// K/Q staged through VGPRs because they are transposed element-wise.
// LDS (dynamic, ~104 KB; MI455X WGP has 320 KB):
//   ks [256][64], vs [64][256], qs [64][64], ps [4][16][256]
// ---------------------------------------------------------------------------
__global__ __launch_bounds__(128) void attn_fused_kernel(
    const __bf16* __restrict__ q, const __bf16* __restrict__ kc,
    const __bf16* __restrict__ vc, __bf16* __restrict__ xout) {
  extern __shared__ __bf16 smem[];
  __bf16* ks = smem;                 // 256*64
  __bf16* vs = ks + 256 * 64;        // 64*256
  __bf16* qs = vs + 64 * 256;        // 64*64
  __bf16* ps = qs + 64 * 64;         // 4*16*256

  const int nblk = blockIdx.x * 64;
  const int h    = blockIdx.y;
  const int b    = blockIdx.z;
  const int tid   = threadIdx.x;
  const int lane  = tid & 31;
  const int wave  = tid >> 5;
  const int lm    = lane & 15;
  const int khalf = lane >> 4;

  // --- stage kc/vc for this (b,h): rows d = hd*H + h, hd = 0..63 ----------
  for (int c = tid; c < 2048; c += 128) {       // 64 hd-rows x 32 8-half chunks
    const int hd  = c >> 5;
    const int off = (c & 31) << 3;
    const size_t grow = ((size_t)b * DD + (size_t)hd * HH + h) * KK + off;
    // vs row copy is contiguous: async DMA straight into LDS (no VGPR data)
    {
      const unsigned dst =
          (unsigned)(size_t)(const void*)(vs + hd * 256 + off);  // LDS offset
      const unsigned long long src = (unsigned long long)(size_t)(vc + grow);
      asm volatile("global_load_async_to_lds_b128 %0, %1, off"
                   :: "v"(dst), "v"(src)
                   : "memory");
    }
    // ks needs an element-wise transpose: VGPR path
    bf16x8 kv = *(const bf16x8*)(kc + grow);
#pragma unroll
    for (int j = 0; j < 8; ++j) ks[(off + j) * 64 + hd] = kv[j];
  }
  // --- stage q tile transposed: qs[n_local][hd] ---------------------------
  for (int c = tid; c < 512; c += 128) {        // 64 hd-rows x 8 chunks
    const int hd   = c >> 3;
    const int noff = (c & 7) << 3;
    bf16x8 qv = *(const bf16x8*)(q + ((size_t)b * DD + (size_t)hd * HH + h) * NN +
                                 nblk + noff);
#pragma unroll
    for (int j = 0; j < 8; ++j) qs[(noff + j) * 64 + hd] = qv[j];
  }
  asm volatile("s_wait_asynccnt 0x0" ::: "memory");  // this wave's DMAs done
  __syncthreads();                                   // all waves' stages done

  // --- scores: 16 n-rows per wave x K=256 cols (16 WMMA tiles) ------------
  f32x8 sacc[16] = {};
#pragma unroll
  for (int kk = 0; kk < 64; kk += 32) {
    bf16x16 af = frag_a(qs + (wave * 16 + lm) * 64 + kk, khalf);
#pragma unroll
    for (int t = 0; t < 16; ++t) {
      bf16x16 bfr =
          *(const bf16x16*)(ks + (t * 16 + lm) * 64 + kk + khalf * 16);
      sacc[t] = wmma_bf16(af, bfr, sacc[t]);
    }
  }

  // --- softmax over K=256 (scale 1/sqrt(64)); rows live in VGPR r,
  //     spread over half-wave lanes (16 katt cols) x 16 tiles --------------
  const float scale = 0.125f;
#pragma unroll
  for (int r = 0; r < 8; ++r) {
    float mx = -3.0e38f;
#pragma unroll
    for (int t = 0; t < 16; ++t) mx = fmaxf(mx, sacc[t][r]);
#pragma unroll
    for (int off = 8; off >= 1; off >>= 1)
      mx = fmaxf(mx, __shfl_xor(mx, off, 16));
    mx *= scale;
    float sum = 0.f;
#pragma unroll
    for (int t = 0; t < 16; ++t) {
      float e = __expf(sacc[t][r] * scale - mx);
      sacc[t][r] = e;
      sum += e;
    }
#pragma unroll
    for (int off = 8; off >= 1; off >>= 1) sum += __shfl_xor(sum, off, 16);
    const float rinv = 1.0f / sum;
#pragma unroll
    for (int t = 0; t < 16; ++t)
      ps[(wave * 16 + khalf * 8 + r) * 256 + t * 16 + lm] =
          f2bf(sacc[t][r] * rinv);
  }
  // per-wave LDS RAW: DS ops are in-order within a wave; keep compiler honest
  asm volatile("" ::: "memory");

  // --- AV: x[n,hd] = prob[n,:] * v[hd,:]^T  (contraction katt=256) --------
  f32x8 xacc[4] = {};
#pragma unroll
  for (int s = 0; s < 8; ++s) {
    bf16x16 af = frag_a(ps + (wave * 16 + lm) * 256 + s * 32, khalf);
#pragma unroll
    for (int t2 = 0; t2 < 4; ++t2) {
      bf16x16 bfr =
          *(const bf16x16*)(vs + (t2 * 16 + lm) * 256 + s * 32 + khalf * 16);
      xacc[t2] = wmma_bf16(af, bfr, xacc[t2]);
    }
  }

  // --- store x[b, d=hd*H+h, n] as bf16 ------------------------------------
#pragma unroll
  for (int t2 = 0; t2 < 4; ++t2) {
    const int hd = t2 * 16 + lm;
#pragma unroll
    for (int r = 0; r < 8; ++r) {
      const int nloc = wave * 16 + khalf * 8 + r;
      xout[((size_t)b * DD + (size_t)hd * HH + h) * NN + nblk + nloc] =
          f2bf(xacc[t2][r]);
    }
  }
}

// ---------------------------------------------------------------------------
extern "C" void kernel_launch(void* const* d_in, const int* in_sizes, int n_in,
                              void* d_out, int out_size, void* d_ws,
                              size_t ws_size, hipStream_t stream) {
  const float* query  = (const float*)d_in[0];
  const float* key    = (const float*)d_in[1];
  const float* value  = (const float*)d_in[2];
  const float* Wq     = (const float*)d_in[3];
  const float* bq     = (const float*)d_in[4];
  const float* Wk     = (const float*)d_in[5];
  const float* bk     = (const float*)d_in[6];
  const float* Wv     = (const float*)d_in[7];
  const float* bv     = (const float*)d_in[8];
  const float* proj_k = (const float*)d_in[9];
  const float* proj_v = (const float*)d_in[10];
  const float* Wm     = (const float*)d_in[11];
  const float* bm     = (const float*)d_in[12];
  float* out = (float*)d_out;

  // workspace partition (bf16 intermediates, ~132 MB total)
  const size_t SZ_BDN = (size_t)BB * DD * NN;  // 16.7M elems
  const size_t SZ_BDK = (size_t)BB * DD * KK;  // 1.05M elems
  __bf16* qb  = (__bf16*)d_ws;
  __bf16* kb_ = qb  + SZ_BDN;
  __bf16* vb  = kb_ + SZ_BDN;
  __bf16* kcb = vb  + SZ_BDN;
  __bf16* vcb = kcb + SZ_BDK;
  __bf16* xb  = vcb + SZ_BDK;

  const long long BDN = (long long)DD * NN;
  const long long BDK = (long long)DD * KK;
  dim3 blk(256);

  // q/k/v = W * x + b  : M=D, Kd=D, Ncol=N, batched over B in grid.z
  dim3 gConv(DD / 128, NN / 128, BB);
  gemm_wmma_kernel<float, float, __bf16, true><<<gConv, blk, 0, stream>>>(
      Wq, 0, query, BDN, bq, qb, BDN, DD, DD, NN);
  gemm_wmma_kernel<float, float, __bf16, true><<<gConv, blk, 0, stream>>>(
      Wk, 0, key, BDN, bk, kb_, BDN, DD, DD, NN);
  gemm_wmma_kernel<float, float, __bf16, true><<<gConv, blk, 0, stream>>>(
      Wv, 0, value, BDN, bv, vb, BDN, DD, DD, NN);

  // sequence compression: kc = k * proj_k, vc = v * proj_v : M=D,Kd=N,Ncol=K
  dim3 gProj(DD / 128, KK / 128, BB);
  gemm_wmma_kernel<__bf16, float, __bf16, false><<<gProj, blk, 0, stream>>>(
      kb_, BDN, proj_k, 0, nullptr, kcb, BDK, DD, NN, KK);
  gemm_wmma_kernel<__bf16, float, __bf16, false><<<gProj, blk, 0, stream>>>(
      vb, BDN, proj_v, 0, nullptr, vcb, BDK, DD, NN, KK);

  // fused scores/softmax/AV
  dim3 gAttn(NN / 64, HH, BB);
  const size_t lds_bytes =
      (size_t)(256 * 64 + 64 * 256 + 64 * 64 + 4 * 16 * 256) * sizeof(__bf16);
  attn_fused_kernel<<<gAttn, dim3(128), lds_bytes, stream>>>(qb, kcb, vcb, xb);

  // out = Wm * x + bm (fp32 output)
  gemm_wmma_kernel<float, __bf16, float, true><<<gConv, blk, 0, stream>>>(
      Wm, 0, xb, BDN, bm, out, BDN, DD, DD, NN);
}
